// DifferentiableAlignmentModule_22849226015003
// MI455X (gfx1250) — matrix operand order repaired and checked by reading, not compiled, not run on hardware
//
#include <hip/hip_runtime.h>
#include <hip/hip_bf16.h>

// Problem constants (from reference)
#define B_     16
#define NTEXT  512
#define TMEL   2048
#define NFEATS 80
#define DIM_   256
#define NCH    512
#define KS     5

typedef __attribute__((ext_vector_type(16))) __bf16 v16bf;
typedef __attribute__((ext_vector_type(8)))  float  v8f;
typedef __attribute__((ext_vector_type(4)))  unsigned int u32x4;

static __device__ __forceinline__ __bf16 f2bf(float f) {
  unsigned u = __builtin_bit_cast(unsigned, f);
  unsigned r = u + 0x7FFFu + ((u >> 16) & 1u);   // round-to-nearest-even
  unsigned short h = (unsigned short)(r >> 16);
  return __builtin_bit_cast(__bf16, h);
}
static __device__ __forceinline__ float bf2f(__bf16 h) {
  unsigned u = ((unsigned)__builtin_bit_cast(unsigned short, h)) << 16;
  return __builtin_bit_cast(float, u);
}
static __device__ __forceinline__ float lrelu(float v) { return v >= 0.0f ? v : 0.1f * v; }

#define WMMA_BF16(A, Bf, C) \
  __builtin_amdgcn_wmma_f32_16x16x32_bf16(false, (A), false, (Bf), (short)0, (C), false, false)

// A fragment: lane holds row (l&15); K contiguous [2*kba, 2*kba+16),
// kba = (l>>4)*8. Two 16B loads (global_load_b128).
static __device__ __forceinline__ v16bf load_frag_a(const __bf16* __restrict__ row, int kba) {
  union { u32x4 q[2]; v16bf v; } u;
  u.q[0] = *(const u32x4*)(row + 2 * kba);
  u.q[1] = *(const u32x4*)(row + 2 * kba + 8);
  return u.v;
}
// B fragment from pre-packed weights: 32 contiguous bytes per lane.
static __device__ __forceinline__ v16bf load_frag_b(const __bf16* __restrict__ p) {
  union { u32x4 q[2]; v16bf v; } u;
  u.q[0] = *(const u32x4*)(p);
  u.q[1] = *(const u32x4*)(p + 8);
  return u.v;
}
static __device__ __forceinline__ v16bf zero_frag() {
  union { u32x4 q[2]; v16bf v; } u;
  u32x4 z = {0u, 0u, 0u, 0u};
  u.q[0] = z; u.q[1] = z;
  return u.v;
}

// ---------------------------------------------------------------------------
// Streaming converts / packs (one-time per launch, bandwidth-bound)
// ---------------------------------------------------------------------------
__global__ void k_zero(float* __restrict__ p, int n)
{
  int i = blockIdx.x * blockDim.x + threadIdx.x;
  if (i < n) p[i] = 0.0f;
}

__global__ void k_cvt(const float* __restrict__ s, __bf16* __restrict__ d, int n)
{
  int i = blockIdx.x * blockDim.x + threadIdx.x;
  if (i < n) d[i] = f2bf(s[i]);
}

// mel [B,80,T] f32 -> melT [B,T,96] bf16 (K padded with zeros)
__global__ void k_melT(const float* __restrict__ mel, __bf16* __restrict__ melT)
{
  int id = blockIdx.x * 256 + threadIdx.x;   // b*2048 + t
  int b = id >> 11, t = id & 2047;
  const float* src = mel + (size_t)b * NFEATS * TMEL + t;
  __bf16* d = melT + (size_t)id * 96;
  for (int f = 0; f < 96; ++f) d[f] = (f < NFEATS) ? f2bf(src[(size_t)f * TMEL]) : f2bf(0.0f);
}

// Pack a [K x N] (generic strides) f32 weight into WMMA-B fragment order:
// dst[((nt*KT + q)*32 + lane)*16 + e] = W[(q*32 + (lane>>4)*16 + e) , nt*16 + (lane&15)]
__global__ void k_packB(const float* __restrict__ src, long sK, long sN,
                        int K, int KT, int Ntiles, __bf16* __restrict__ dst)
{
  int id = blockIdx.x * blockDim.x + threadIdx.x;
  if (id >= Ntiles * KT * 32) return;
  int l = id & 31;
  int q = (id >> 5) % KT;
  int nt = id / (32 * KT);
  int n = nt * 16 + (l & 15);
  int kbase = q * 32 + ((l >> 4) * 16);
  __bf16* d = dst + (size_t)id * 16;
  for (int e = 0; e < 16; ++e) {
    int kk = kbase + e;
    float v = (kk < K) ? src[(size_t)kk * sK + (size_t)n * sN] : 0.0f;
    d[e] = f2bf(v);
  }
}

// ---------------------------------------------------------------------------
// Kernel: x @ W + b. A = xbf [8192,256] bf16, B = packed [16 ntiles, KT=8].
// BF16OUT: write bf16 (x_key to ws) or f32 (x_value to d_out).
// ---------------------------------------------------------------------------
template <bool BF16OUT>
__global__ void k_gemm_x(const __bf16* __restrict__ A, const __bf16* __restrict__ Bp,
                         const float* __restrict__ bias, void* __restrict__ outp)
{
  int wid = blockIdx.x * (blockDim.x >> 5) + (threadIdx.x >> 5);
  int mt = wid >> 4, nt = wid & 15;
  int l = threadIdx.x & 31;
  int kba = (l >> 4) * 8;
  const __bf16* arow = A + (size_t)(mt * 16 + (l & 15)) * DIM_;
  const __bf16* bp   = Bp + ((size_t)nt * 8) * 512 + (size_t)l * 16;

  v8f acc = {0,0,0,0,0,0,0,0};
#pragma unroll
  for (int q = 0; q < 8; ++q) {
    v16bf a = load_frag_a(arow + q * 32, kba);
    v16bf b = load_frag_b(bp + q * 512);
    acc = WMMA_BF16(a, b, acc);
  }
  int bn = nt * 16 + (l & 15);
  float bb = bias[bn];
#pragma unroll
  for (int i = 0; i < 8; ++i) {
    size_t idx = (size_t)(mt * 16 + i + 8 * (l >> 4)) * DIM_ + bn;
    if (BF16OUT) ((__bf16*)outp)[idx] = f2bf(acc[i] + bb);
    else         ((float*)outp)[idx]  = acc[i] + bb;
  }
}

// ---------------------------------------------------------------------------
// Kernel: h0[b,t,c] = lrelu(melT[b,t,:] @ Wpre + bpre).  K=96 (KT=3), N=512.
// ---------------------------------------------------------------------------
__global__ void k_pre(const __bf16* __restrict__ melT, const __bf16* __restrict__ Bp,
                      const float* __restrict__ bp, __bf16* __restrict__ h)
{
  int wid = blockIdx.x * (blockDim.x >> 5) + (threadIdx.x >> 5);
  int ct = wid & 31; int tt = (wid >> 5) & 127; int b = wid >> 12;
  int l = threadIdx.x & 31;
  int t0 = tt * 16, c0 = ct * 16;
  int kba = (l >> 4) * 8;
  const __bf16* arow = melT + (size_t)(b * TMEL + t0 + (l & 15)) * 96;
  const __bf16* bpp  = Bp + ((size_t)ct * 3) * 512 + (size_t)l * 16;

  v8f acc = {0,0,0,0,0,0,0,0};
#pragma unroll
  for (int q = 0; q < 3; ++q) {
    v16bf a = load_frag_a(arow + q * 32, kba);
    v16bf bf = load_frag_b(bpp + q * 512);
    acc = WMMA_BF16(a, bf, acc);
  }
  int c = c0 + (l & 15);
  float bb = bp[c];
#pragma unroll
  for (int i = 0; i < 8; ++i) {
    int t = t0 + i + 8 * (l >> 4);
    h[(size_t)(b * TMEL + t) * NCH + c] = f2bf(lrelu(acc[i] + bb));
  }
}

// ---------------------------------------------------------------------------
// Kernel: conv layer. h stored [B,T,C] bf16 (K=C contiguous). Per tap k:
// GEMM with shifted A rows; 5 taps x KT=16 accumulated. Residual from bf16 h.
// ---------------------------------------------------------------------------
__global__ void k_conv(const __bf16* __restrict__ hin, const __bf16* __restrict__ pL,
                       const float* __restrict__ bias, __bf16* __restrict__ hout)
{
  int wid = blockIdx.x * (blockDim.x >> 5) + (threadIdx.x >> 5);
  int ct = wid & 31; int tt = (wid >> 5) & 127; int b = wid >> 12;
  int l = threadIdx.x & 31;
  int t0 = tt * 16, c0 = ct * 16;
  int m = l & 15;
  int kba = (l >> 4) * 8;
  const v16bf zf = zero_frag();

  v8f acc = {0,0,0,0,0,0,0,0};
  for (int k = 0; k < KS; ++k) {
    int ts = t0 + m + k - 2;
    bool inb = (ts >= 0) && (ts < TMEL);
    const __bf16* arow = hin + (size_t)(b * TMEL + ts) * NCH;
    const __bf16* bpp  = pL + (size_t)k * (NCH * NCH) + ((size_t)ct * 16) * 512 + (size_t)l * 16;
    if (k + 1 < KS) __builtin_prefetch(pL + (size_t)(k + 1) * (NCH * NCH) + ((size_t)ct * 16) * 512 + (size_t)l * 16, 0, 1);
#pragma unroll 4
    for (int q = 0; q < 16; ++q) {
      v16bf a = inb ? load_frag_a(arow + q * 32, kba) : zf;
      v16bf bf = load_frag_b(bpp + q * 512);
      acc = WMMA_BF16(a, bf, acc);
    }
  }
  int c = c0 + (l & 15);
  float bb = bias[c];
#pragma unroll
  for (int i = 0; i < 8; ++i) {
    int t = t0 + i + 8 * (l >> 4);
    size_t idx = (size_t)(b * TMEL + t) * NCH + c;
    hout[idx] = f2bf(bf2f(hin[idx]) + lrelu(acc[i] + bb));
  }
}

// ---------------------------------------------------------------------------
// Kernel: mel_h[b,t,d] = h[b,t,:] @ Wproj + bproj.  K=512 (KT=16), N=256.
// ---------------------------------------------------------------------------
__global__ void k_proj(const __bf16* __restrict__ h, const __bf16* __restrict__ Bp,
                       const float* __restrict__ bp, __bf16* __restrict__ melh)
{
  int wid = blockIdx.x * (blockDim.x >> 5) + (threadIdx.x >> 5);
  int dt = wid & 15; int tt = (wid >> 4) & 127; int b = wid >> 11;
  int l = threadIdx.x & 31;
  int t0 = tt * 16, d0 = dt * 16;
  int kba = (l >> 4) * 8;
  const __bf16* arow = h + (size_t)(b * TMEL + t0 + (l & 15)) * NCH;
  const __bf16* bpp  = Bp + ((size_t)dt * 16) * 512 + (size_t)l * 16;

  v8f acc = {0,0,0,0,0,0,0,0};
#pragma unroll 4
  for (int q = 0; q < 16; ++q) {
    v16bf a = load_frag_a(arow + q * 32, kba);
    v16bf bf = load_frag_b(bpp + q * 512);
    acc = WMMA_BF16(a, bf, acc);
  }
  int d = d0 + (l & 15);
  float bb = bp[d];
#pragma unroll
  for (int i = 0; i < 8; ++i) {
    int t = t0 + i + 8 * (l >> 4);
    melh[(size_t)(b * TMEL + t) * DIM_ + d] = f2bf(acc[i] + bb);
  }
}

// ---------------------------------------------------------------------------
// Kernel: fused attention. energies = melh @ xkey^T (WMMA; xkey rows ARE the
// B-fragments since K is contiguous) -> scale * e_weight -> key mask -> row
// softmax over N=512 in 32KB LDS -> alpha [B,N,T] + duration atomics.
// Block = 256 thr (8 waves), one per (b, 16-frame tile); wave w owns n-tiles [4w,4w+4).
// ---------------------------------------------------------------------------
__global__ void k_attn(const __bf16* __restrict__ melh, const __bf16* __restrict__ xkey,
                       const float* __restrict__ ew, const int* __restrict__ xlen,
                       const int* __restrict__ mlen,
                       float* __restrict__ alpha, float* __restrict__ dur)
{
  __shared__ float sm[16 * NTEXT];
  __shared__ float red[256];
  __shared__ float invs[16];
  int b = blockIdx.y, t0 = blockIdx.x * 16;
  int wv = threadIdx.x >> 5, l = threadIdx.x & 31;
  int xl = xlen[b], ml = mlen[b];
  int nn = l & 15;
  int kba = (l >> 4) * 8, kbb = (l >> 4) * 16;
  const __bf16* arow = melh + (size_t)(b * TMEL + t0 + (l & 15)) * DIM_;

  v8f acc[4] = {};
#pragma unroll
  for (int q = 0; q < 8; ++q) {
    v16bf a = load_frag_a(arow + q * 32, kba);
#pragma unroll
    for (int u = 0; u < 4; ++u) {
      int n0 = (wv * 4 + u) * 16;
      const __bf16* krow = xkey + (size_t)(b * NTEXT + n0 + nn) * DIM_ + q * 32 + kbb;
      v16bf bf = load_frag_b(krow);
      acc[u] = WMMA_BF16(a, bf, acc[u]);
    }
  }
#pragma unroll
  for (int u = 0; u < 4; ++u) {
    int n = (wv * 4 + u) * 16 + nn;
#pragma unroll
    for (int i = 0; i < 8; ++i) {
      int row = i + 8 * (l >> 4);
      float v = acc[u][i] * 0.0625f;   // 1/sqrt(256)
      v *= ew[((size_t)b * NTEXT + n) * TMEL + (t0 + row)];
      if (n >= xl) v = -INFINITY;
      sm[row * NTEXT + n] = v;
    }
  }
  __syncthreads();

  int r = threadIdx.x >> 4, c = threadIdx.x & 15;
  float mx = -INFINITY;
  for (int n = c; n < NTEXT; n += 16) mx = fmaxf(mx, sm[r * NTEXT + n]);
  red[threadIdx.x] = mx;
  __syncthreads();
  if (c == 0) {
    float M = red[r * 16];
    for (int j = 1; j < 16; ++j) M = fmaxf(M, red[r * 16 + j]);
    red[r * 16] = M;
  }
  __syncthreads();
  float M = red[r * 16];
  __syncthreads();
  float s = 0.0f;
  for (int n = c; n < NTEXT; n += 16) {
    float e = __expf(sm[r * NTEXT + n] - M);
    sm[r * NTEXT + n] = e;
    s += e;
  }
  red[threadIdx.x] = s;
  __syncthreads();
  if (c == 0) {
    float S = 0.0f;
    for (int j = 0; j < 16; ++j) S += red[r * 16 + j];
    invs[r] = ((t0 + r) < ml) ? (1.0f / S) : 0.0f;
  }
  __syncthreads();

  float inv = invs[r];
  int t = t0 + r;
  for (int n = c; n < NTEXT; n += 16)
    alpha[((size_t)b * NTEXT + n) * TMEL + t] = sm[r * NTEXT + n] * inv;

  for (int n = threadIdx.x; n < NTEXT; n += 256) {
    float p = 0.0f;
#pragma unroll
    for (int rr = 0; rr < 16; ++rr) p += sm[rr * NTEXT + n] * invs[rr];
    atomicAdd(dur + (size_t)b * NTEXT + n, p);
  }
}

// ---------------------------------------------------------------------------
__global__ void k_pos(const float* __restrict__ dur, float* __restrict__ epos)
{
  int b = blockIdx.x;
  if (threadIdx.x == 0) {
    float csum = 0.0f;
    for (int n = 0; n < NTEXT; ++n) {
      float d = dur[(size_t)b * NTEXT + n];
      csum += d;
      epos[(size_t)b * NTEXT + n] = csum - 0.5f * d;
    }
  }
}

__global__ void k_rec(const float* __restrict__ epos, const int* __restrict__ xlen,
                      const int* __restrict__ mlen, float* __restrict__ rec)
{
  __shared__ float es[NTEXT];
  int b = blockIdx.y;
  int t = blockIdx.x * 256 + threadIdx.x;
  for (int n = threadIdx.x; n < NTEXT; n += 256) es[n] = epos[(size_t)b * NTEXT + n];
  __syncthreads();
  int xl = xlen[b], ml = mlen[b];
  bool tv = (t < ml);
  float q = tv ? (float)t : 0.0f;
  float M = -INFINITY;
  for (int n = 0; n < xl; ++n) { float d = q - es[n]; M = fmaxf(M, -0.2f * d * d); }
  float S = 0.0f;
  for (int n = 0; n < xl; ++n) { float d = q - es[n]; S += __expf(-0.2f * d * d - M); }
  float inv = tv ? (1.0f / S) : 0.0f;
  for (int n = 0; n < NTEXT; ++n) {
    float v = 0.0f;
    if (n < xl) { float d = q - es[n]; v = __expf(-0.2f * d * d - M) * inv; }
    rec[((size_t)b * NTEXT + n) * TMEL + t] = v;
  }
}

// ---------------------------------------------------------------------------
extern "C" void kernel_launch(void* const* d_in, const int* in_sizes, int n_in,
                              void* d_out, int out_size, void* d_ws, size_t ws_size,
                              hipStream_t stream)
{
  const float* x     = (const float*)d_in[0];
  const int*   xlen  = (const int*)  d_in[1];
  const float* mel   = (const float*)d_in[2];
  const int*   mlen  = (const int*)  d_in[3];
  const float* ew    = (const float*)d_in[4];
  const float* Wk    = (const float*)d_in[5];
  const float* bk    = (const float*)d_in[6];
  const float* Wv    = (const float*)d_in[7];
  const float* bv    = (const float*)d_in[8];
  const float* Wpre  = (const float*)d_in[9];
  const float* bpre  = (const float*)d_in[10];
  const float* convw = (const float*)d_in[11];
  const float* convb = (const float*)d_in[12];
  const float* Wproj = (const float*)d_in[13];
  const float* bproj = (const float*)d_in[14];

  // d_out: x_value | durations | alpha | reconst_alpha
  float* out  = (float*)d_out;
  float* xval = out;
  float* dur  = out + (size_t)B_ * NTEXT * DIM_;
  float* alph = dur + (size_t)B_ * NTEXT;
  float* rec  = alph + (size_t)B_ * NTEXT * TMEL;

  // workspace: epos (f32) then bf16 region
  float*  epos   = (float*)d_ws;
  __bf16* bfb    = (__bf16*)((char*)d_ws + (size_t)B_ * NTEXT * sizeof(float));
  __bf16* xbf    = bfb;                                     // 2,097,152
  __bf16* xkeybf = xbf    + (size_t)2097152;                // 2,097,152
  __bf16* melT   = xkeybf + (size_t)2097152;                // 3,145,728
  __bf16* hA     = melT   + (size_t)3145728;                // 16,777,216
  __bf16* hB     = hA     + (size_t)16777216;               // 16,777,216
  __bf16* melhbf = hB     + (size_t)16777216;               // 8,388,608
  __bf16* pWk    = melhbf + (size_t)8388608;                // 65,536
  __bf16* pWv    = pWk    + (size_t)65536;                  // 65,536
  __bf16* pWpre  = pWv    + (size_t)65536;                  // 49,152
  __bf16* pWproj = pWpre  + (size_t)49152;                  // 131,072
  __bf16* pConv  = pWproj + (size_t)131072;                 // 15 * 262,144

  // 0) init + converts + packs
  k_zero<<<(B_ * NTEXT + 255) / 256, 256, 0, stream>>>(dur, B_ * NTEXT);
  k_cvt<<<8192, 256, 0, stream>>>(x, xbf, B_ * NTEXT * DIM_);
  k_melT<<<(B_ * TMEL) / 256, 256, 0, stream>>>(mel, melT);
  k_packB<<<16, 256, 0, stream>>>(Wk,    DIM_, 1, DIM_, 8,  16, pWk);
  k_packB<<<16, 256, 0, stream>>>(Wv,    DIM_, 1, DIM_, 8,  16, pWv);
  k_packB<<<12, 256, 0, stream>>>(Wpre,  NCH,  1, NFEATS, 3, 32, pWpre);
  k_packB<<<32, 256, 0, stream>>>(Wproj, DIM_, 1, NCH, 16, 16, pWproj);
  for (int lyr = 0; lyr < 3; ++lyr)
    for (int k = 0; k < KS; ++k)
      k_packB<<<64, 256, 0, stream>>>(convw + (size_t)lyr * NCH * NCH * KS + k,
                                      KS, (long)NCH * KS, NCH, 16, 32,
                                      pConv + (size_t)(lyr * KS + k) * NCH * NCH);

  // 1) x_key (bf16 ws) / x_value (f32 out): 8192 tiles each, 4 waves/block
  k_gemm_x<true ><<<2048, 128, 0, stream>>>(xbf, pWk, bk, xkeybf);
  k_gemm_x<false><<<2048, 128, 0, stream>>>(xbf, pWv, bv, xval);

  // 2) pre-projection: 65536 waves
  k_pre<<<16384, 128, 0, stream>>>(melT, pWpre, bpre, hA);

  // 3) conv stack (ping-pong)
  k_conv<<<16384, 128, 0, stream>>>(hA, pConv + (size_t)0 * KS * NCH * NCH, convb + 0 * NCH, hB);
  k_conv<<<16384, 128, 0, stream>>>(hB, pConv + (size_t)1 * KS * NCH * NCH, convb + 1 * NCH, hA);
  k_conv<<<16384, 128, 0, stream>>>(hA, pConv + (size_t)2 * KS * NCH * NCH, convb + 2 * NCH, hB);

  // 4) projection: 32768 waves
  k_proj<<<8192, 128, 0, stream>>>(hB, pWproj, bproj, melhbf);

  // 5) fused attention + softmax + alpha + durations
  k_attn<<<dim3(TMEL / 16, B_), 256, 0, stream>>>(melhbf, xkeybf, ew, xlen, mlen, alph, dur);

  // 6) aligned positions
  k_pos<<<B_, 32, 0, stream>>>(dur, epos);

  // 7) Gaussian reconstruction softmax
  k_rec<<<dim3(TMEL / 256, B_), 256, 0, stream>>>(epos, xlen, mlen, rec);
}
